// GMEdgeNet_63333587747356
// MI455X (gfx1250) — matrix operand order repaired and checked by reading, not compiled, not run on hardware
//
#include <hip/hip_runtime.h>
#include <float.h>
#include <limits.h>

// ---------------------------------------------------------------------------
// GMEdgeNet forward for MI455X (gfx1250), wave32 + WMMA f16->f32.
//
//  * edge first layer:  [x_i, x_j-x_i]@W1 = x_i@(W1a-W1b) + x_j@W1b
//    -> per-node P,Q GEMMs instead of per-edge 256-K GEMM.
//  * segment_max via ordered-int encoding + GLOBAL_ATOMIC_MAX_I32.
//  * global-pool term of final layer folded into a 384-wide bias vector.
//  * all weights pre-transposed to [N,K] so B fragments are contiguous
//    (2x global_load_b128 per fragment); A tiles staged in LDS in WMMA
//    fragment order (2x ds_load_b128 per fragment, no repacking movs).
// ---------------------------------------------------------------------------

typedef __attribute__((ext_vector_type(16))) _Float16 v16h;
typedef __attribute__((ext_vector_type(4)))  _Float16 v4h;
typedef __attribute__((ext_vector_type(8)))  float    v8f;
typedef __attribute__((ext_vector_type(4)))  float    v4f;
typedef __attribute__((ext_vector_type(4)))  int      v4i;

#define GM_N      50000
#define GM_E      800000
#define GM_HID    128
#define GM_OUT    384

// ---- ordered-int float max encoding (monotonic f32 <-> i32) ----------------
__device__ __forceinline__ int gm_enc(float f) {
  int i = __float_as_int(f);
  return (i < 0) ? (i ^ 0x7fffffff) : i;
}
__device__ __forceinline__ float gm_dec(int i) {
  if (i == INT_MIN) return 0.0f;   // empty segment -> 0 (PyG fill)
  return __int_as_float((i < 0) ? (i ^ 0x7fffffff) : i);
}

// (row-k) -> (fragment hi, h) mapping: contiguous k = 16*hi + h, same order as the B loads.
__device__ __forceinline__ void gm_kmap(int c, int& hi, int& h) {
  hi = c >> 4; h = c & 15;
}

// ---- small helpers ----------------------------------------------------------
__global__ void gm_fill_i32(int* __restrict__ p, int v, int n) {
  int i = blockIdx.x * blockDim.x + threadIdx.x;
  if (i < n) p[i] = v;
}

// f32 [K,N] row-major  ->  f16 [N,K] (transposed, B-fragment friendly)
__global__ void gm_cvt_f16_t(const float* __restrict__ s, _Float16* __restrict__ d,
                             int K, int N) {
  int i = blockIdx.x * blockDim.x + threadIdx.x;
  if (i >= K * N) return;
  int k = i / N, n = i - k * N;
  d[n * K + k] = (_Float16)s[i];
}

// W1 [2K x N] -> transposed f16 Wdiff = (W1a-W1b)^T, Wb = W1b^T, each [N,K]
__global__ void gm_wdiff_t(const float* __restrict__ W1, _Float16* __restrict__ Wdiff,
                           _Float16* __restrict__ Wb, int K, int N) {
  int i = blockIdx.x * blockDim.x + threadIdx.x;
  if (i >= K * N) return;
  int k = i / N, n = i - k * N;
  float a = W1[i], b = W1[i + K * N];
  Wdiff[n * K + k] = (_Float16)(a - b);
  Wb[n * K + k]    = (_Float16)b;
}

// conv1 P/Q: tiny K=3 first layer, pure VALU (not worth WMMA).
__global__ void gm_pq3(const float* __restrict__ x, const float* __restrict__ W1,
                       float* __restrict__ P, float* __restrict__ Q) {
  int idx = blockIdx.x * blockDim.x + threadIdx.x;
  if (idx >= GM_N * GM_HID) return;
  int n = idx >> 7, f = idx & 127;
  float x0 = x[n * 3 + 0], x1 = x[n * 3 + 1], x2 = x[n * 3 + 2];
  float wa0 = W1[0 * 128 + f], wa1 = W1[1 * 128 + f], wa2 = W1[2 * 128 + f];
  float wb0 = W1[3 * 128 + f], wb1 = W1[4 * 128 + f], wb2 = W1[5 * 128 + f];
  P[idx] = x0 * (wa0 - wb0) + x1 * (wa1 - wb1) + x2 * (wa2 - wb2);
  Q[idx] = x0 * wb0 + x1 * wb1 + x2 * wb2;
}

// ---- generic WMMA GEMM: Y[M,N] = act(A[M,K] @ W[K,N] + bias) ---------------
// W passed transposed as Wt[N,K]. Block = 256 thr = 8 wave32; block tile is
// 32 rows x 128 cols: each wave owns one 16-col strip and two 16-row A
// fragments sharing one B fragment (2 WMMA per B load).
// AMODE: 0 = f32 A, 1 = ordered-int encoded A (fused segment-max decode).
template <int AMODE>
__global__ void gm_wmma_gemm(const void* __restrict__ Av, int lda,
                             const _Float16* __restrict__ Wt,   // [N,K]
                             const float* __restrict__ bias,    // may be null
                             float* __restrict__ Y, int ldy,
                             int M, int N, int K, int relu) {
  __shared__ alignas(32) _Float16 As[2][32][16];   // [mtile][lane][h], fragment order
  const int tid   = threadIdx.x;
  const int lane  = tid & 31;
  const int mbase = blockIdx.x * 32;
  const int nbase = blockIdx.y * 128 + (tid >> 5) * 16;
  const int ncol  = nbase + (lane & 15);
  const int hiK   = (lane >> 4) & 1;
  const _Float16* wcol = Wt + ncol * K + hiK * 16;

  // per-thread staging coords: one float4 (4 k-values of one row) per k-step
  const int sr = tid >> 3;          // 0..31: row within 32-row tile
  const int sc = (tid & 7) * 4;     // 0..28: k-offset within 32-k step
  int shi, shb; gm_kmap(sc, shi, shb);
  _Float16* sdst = &As[sr >> 4][(sr & 15) + shi * 16][shb];
  int arow = mbase + sr; if (arow >= M) arow = M - 1;   // clamp tail reads
  const int abase = arow * lda;

  v8f acc0 = {}, acc1 = {};
  for (int k0 = 0; k0 < K; k0 += 32) {
    __syncthreads();                                   // WAR vs previous reads
    v4h hv;
    if (AMODE == 0) {
      v4f a = *(const v4f*)((const float*)Av + abase + k0 + sc);
      hv[0] = (_Float16)a[0]; hv[1] = (_Float16)a[1];
      hv[2] = (_Float16)a[2]; hv[3] = (_Float16)a[3];
    } else {
      v4i a = *(const v4i*)((const int*)Av + abase + k0 + sc);
      hv[0] = (_Float16)gm_dec(a[0]); hv[1] = (_Float16)gm_dec(a[1]);
      hv[2] = (_Float16)gm_dec(a[2]); hv[3] = (_Float16)gm_dec(a[3]);
    }
    *(v4h*)sdst = hv;
    __syncthreads();
    v16h b  = *(const v16h*)(wcol + k0);
    v16h a0 = *(const v16h*)(&As[0][lane][0]);
    v16h a1 = *(const v16h*)(&As[1][lane][0]);
    acc0 = __builtin_amdgcn_wmma_f32_16x16x32_f16(false, a0, false, b,
                                                  (short)0, acc0, false, false);
    acc1 = __builtin_amdgcn_wmma_f32_16x16x32_f16(false, a1, false, b,
                                                  (short)0, acc1, false, false);
  }
  float bb = bias ? bias[ncol] : 0.0f;
#pragma unroll
  for (int v = 0; v < 8; ++v) {                       // f32 C/D layout
    int m0 = mbase + v + hiK * 8;
    int m1 = m0 + 16;
    float r0 = acc0[v] + bb, r1 = acc1[v] + bb;
    if (relu) { r0 = fmaxf(r0, 0.0f); r1 = fmaxf(r1, 0.0f); }
    if (m0 < M) Y[m0 * ldy + ncol] = r0;
    if (m1 < M) Y[m1 * ldy + ncol] = r1;
  }
}

// ---- edge message + segment max --------------------------------------------
// Block handles 32 edges: h = relu(P[dst]+Q[src]+b1) staged to LDS in WMMA
// fragment order (f16), then 8 waves compute h @ W2 (K=128) with WMMA
// (2 edge-tiles share each B fragment), add b2, atomic-max into AGG[dst].
__global__ void gm_edge_msg(const int* __restrict__ src, const int* __restrict__ dst,
                            const float* __restrict__ P, const float* __restrict__ Q,
                            const float* __restrict__ b1,
                            const _Float16* __restrict__ W2t,   // [128,128] transposed
                            const float* __restrict__ b2,
                            int* __restrict__ agg, int ldA, int colOff) {
  __shared__ alignas(32) _Float16 Hs[2][4][32][16];   // [etile][kblk][lane][h]
  __shared__ int ds[32];
  const int ebase = blockIdx.x * 32;
  const int tid   = threadIdx.x;
  if (tid < 32) ds[tid] = dst[ebase + tid];
  __syncthreads();
  for (int q = tid; q < 1024; q += 256) {             // 32 edges x 32 float4s
    int e = q >> 5;
    int f = (q & 31) * 4;
    int s = src[ebase + e];
    int d = ds[e];
    v4f pv = *(const v4f*)(P + d * 128 + f);
    v4f qv = *(const v4f*)(Q + s * 128 + f);
    v4f bv = *(const v4f*)(b1 + f);
    v4h hv;
#pragma unroll
    for (int j = 0; j < 4; ++j) hv[j] = (_Float16)fmaxf(pv[j] + qv[j] + bv[j], 0.0f);
    int c = f & 31, hi, hb; gm_kmap(c, hi, hb);
    *(v4h*)&Hs[e >> 4][f >> 5][(e & 15) + hi * 16][hb] = hv;
  }
  __syncthreads();
  const int lane = tid & 31;
  const int ncol = (tid >> 5) * 16 + (lane & 15);
  const int hiK  = (lane >> 4) & 1;
  const _Float16* wcol = W2t + ncol * 128 + hiK * 16;
  v8f acc0 = {}, acc1 = {};
#pragma unroll
  for (int kb = 0; kb < 4; ++kb) {
    v16h b  = *(const v16h*)(wcol + kb * 32);
    v16h a0 = *(const v16h*)(&Hs[0][kb][lane][0]);
    v16h a1 = *(const v16h*)(&Hs[1][kb][lane][0]);
    acc0 = __builtin_amdgcn_wmma_f32_16x16x32_f16(false, a0, false, b,
                                                  (short)0, acc0, false, false);
    acc1 = __builtin_amdgcn_wmma_f32_16x16x32_f16(false, a1, false, b,
                                                  (short)0, acc1, false, false);
  }
  float bb = b2[ncol];
#pragma unroll
  for (int v = 0; v < 8; ++v) {
    int e0 = v + hiK * 8;
    int e1 = e0 + 16;
    atomicMax(&agg[ds[e0] * ldA + colOff + ncol], gm_enc(acc0[v] + bb));
    atomicMax(&agg[ds[e1] * ldA + colOff + ncol], gm_enc(acc1[v] + bb));
  }
}

// ---- global max pooling over X123 columns ----------------------------------
__global__ void gm_colmax(const float* __restrict__ X, int ldx, int M, int* __restrict__ gmax) {
  int c = threadIdx.x;                 // blockDim.x == 384
  float m = -FLT_MAX;
  for (int r = blockIdx.x; r < M; r += gridDim.x)
    m = fmaxf(m, X[r * ldx + c]);
  atomicMax(&gmax[c], gm_enc(m));
}

// biasF[n] = b1f[n] + sum_k g[k] * W1f[384+k][n]   (pool term folded to bias)
__global__ void gm_gvec(const int* __restrict__ gmax, const float* __restrict__ W1f,
                        const float* __restrict__ b1f, float* __restrict__ biasF) {
  int n = threadIdx.x;                 // blockDim.x == 384
  float s = b1f[n];
  for (int k = 0; k < GM_OUT; ++k)
    s += gm_dec(gmax[k]) * W1f[(GM_OUT + k) * GM_OUT + n];
  biasF[n] = s;
}

// ---------------------------------------------------------------------------
extern "C" void kernel_launch(void* const* d_in, const int* in_sizes, int n_in,
                              void* d_out, int out_size, void* d_ws, size_t ws_size,
                              hipStream_t stream) {
  (void)in_sizes; (void)n_in; (void)out_size; (void)ws_size;

  // ---- inputs (setup_inputs dict order, params flattened in insertion order)
  const float* x      = (const float*)d_in[0];
  const int*   meshEI = (const int*)d_in[1];   // [2,E]: row0=src, row1=dst
  const int*   geoEI  = (const int*)d_in[2];
  const float* p[40];
  for (int i = 0; i < 40; ++i) p[i] = (const float*)d_in[3 + i];
  // conv c (0..2): base=c*12; mesh W1=+0 b1=+1 W2=+2 b2=+3; geo +4..7; comb +8..11
  const float* fW1 = p[36]; const float* fb1 = p[37];
  const float* fW2 = p[38]; const float* fb2 = p[39];

  // ---- workspace arena ------------------------------------------------------
  char* w = (char*)d_ws;
  size_t off = 0;
  auto alloc = [&](size_t bytes) -> void* {
    void* q = w + off; off += (bytes + 255) & ~(size_t)255; return q;
  };
  float* X123 = (float*)alloc((size_t)GM_N * 384 * 4);       // x1|x2|x3 columns
  int*   AGG  = (int*)  alloc((size_t)GM_N * 256 * 4);       // mesh|geo encoded max
  float* P    = (float*)alloc((size_t)GM_N * 128 * 4);
  float* Q    = (float*)alloc((size_t)GM_N * 128 * 4);
  float* T1   = (float*)alloc((size_t)GM_N * 128 * 4);
  float* T2   = (float*)AGG;                                  // reuse AGG+P after conv3
  int*   gmax = (int*)  alloc(384 * 4);
  float* biasF= (float*)alloc(384 * 4);
  // f16 weights (all stored TRANSPOSED: [N,K])
  _Float16* Wdiff[2][2]; _Float16* Wb[2][2];                  // [conv2/3][mesh/geo]
  for (int c = 0; c < 2; ++c) for (int t = 0; t < 2; ++t) {
    Wdiff[c][t] = (_Float16*)alloc(128 * 128 * 2);
    Wb[c][t]    = (_Float16*)alloc(128 * 128 * 2);
  }
  _Float16* W2t[3][2];                                        // edge 2nd layer
  for (int c = 0; c < 3; ++c) for (int t = 0; t < 2; ++t)
    W2t[c][t] = (_Float16*)alloc(128 * 128 * 2);
  _Float16* cW1t[3]; _Float16* cW2t[3];
  for (int c = 0; c < 3; ++c) { cW1t[c] = (_Float16*)alloc(256 * 128 * 2);
                                cW2t[c] = (_Float16*)alloc(128 * 128 * 2); }
  _Float16* fW1t = (_Float16*)alloc(384 * 384 * 2);           // rows 0..383 of final W1
  _Float16* fW2t = (_Float16*)alloc(384 * 384 * 2);

  // ---- weight conversions (transpose to [N,K] f16) --------------------------
  const int KN = 128 * 128;
  const dim3 B256(256);
  for (int c = 0; c < 2; ++c) for (int t = 0; t < 2; ++t)
    gm_wdiff_t<<<(KN + 255) / 256, B256, 0, stream>>>(p[(c + 1) * 12 + t * 4 + 0],
                                                      Wdiff[c][t], Wb[c][t], 128, 128);
  for (int c = 0; c < 3; ++c) for (int t = 0; t < 2; ++t)
    gm_cvt_f16_t<<<(KN + 255) / 256, B256, 0, stream>>>(p[c * 12 + t * 4 + 2],
                                                        W2t[c][t], 128, 128);
  for (int c = 0; c < 3; ++c) {
    gm_cvt_f16_t<<<(2 * KN + 255) / 256, B256, 0, stream>>>(p[c * 12 + 8], cW1t[c], 256, 128);
    gm_cvt_f16_t<<<(KN + 255) / 256, B256, 0, stream>>>(p[c * 12 + 10], cW2t[c], 128, 128);
  }
  gm_cvt_f16_t<<<(384 * 384 + 255) / 256, B256, 0, stream>>>(fW1, fW1t, 384, 384);
  gm_cvt_f16_t<<<(384 * 384 + 255) / 256, B256, 0, stream>>>(fW2, fW2t, 384, 384);

  const int MT = (GM_N + 31) / 32;              // 1563 row tiles (tail guarded)
  auto gemm_f32 = [&](const float* A, int lda, const _Float16* Wt, const float* bias,
                      float* Y, int ldy, int N, int K, int relu) {
    gm_wmma_gemm<0><<<dim3(MT, N / 128), B256, 0, stream>>>(A, lda, Wt, bias, Y, ldy,
                                                            GM_N, N, K, relu);
  };

  // ---- three GM-EdgeConv layers --------------------------------------------
  for (int c = 0; c < 3; ++c) {
    const float* xin = (c == 0) ? nullptr : X123 + (size_t)(c - 1) * 128;
    gm_fill_i32<<<(GM_N * 256 + 255) / 256, B256, 0, stream>>>(AGG, INT_MIN, GM_N * 256);
    for (int t = 0; t < 2; ++t) {               // t: 0=mesh, 1=geo
      const float* b1 = p[c * 12 + t * 4 + 1];
      const float* b2 = p[c * 12 + t * 4 + 3];
      if (c == 0) {
        gm_pq3<<<(GM_N * 128 + 255) / 256, B256, 0, stream>>>(x, p[t * 4 + 0], P, Q);
      } else {
        gemm_f32(xin, 384, Wdiff[c - 1][t], nullptr, P, 128, 128, 128, 0);
        gemm_f32(xin, 384, Wb[c - 1][t],    nullptr, Q, 128, 128, 128, 0);
      }
      const int* ei = (t == 0) ? meshEI : geoEI;
      gm_edge_msg<<<GM_E / 32, B256, 0, stream>>>(ei, ei + GM_E, P, Q, b1,
                                                  W2t[c][t], b2, AGG, 256, t * 128);
    }
    // comb MLP: decode(AGG)[N,256] -> relu GEMM -> GEMM -> X123 column block c
    gm_wmma_gemm<1><<<dim3(MT, 1), B256, 0, stream>>>(AGG, 256, cW1t[c], p[c * 12 + 9],
                                                      T1, 128, GM_N, 128, 256, 1);
    gemm_f32(T1, 128, cW2t[c], p[c * 12 + 11], X123 + (size_t)c * 128, 384, 128, 128, 0);
  }

  // ---- global pool + final MLP ---------------------------------------------
  gm_fill_i32<<<2, B256, 0, stream>>>(gmax, INT_MIN, 384);
  gm_colmax<<<256, 384, 0, stream>>>(X123, 384, GM_N, gmax);
  gm_gvec<<<1, 384, 0, stream>>>(gmax, fW1, fb1, biasF);
  gemm_f32(X123, 384, fW1t, biasF, T2, 384, 384, 384, 1);
  gemm_f32(T2, 384, fW2t, fb2, (float*)d_out, 384, 384, 384, 0);
}
